// RnnWithDones_39419209843112
// MI455X (gfx1250) — matrix operand order repaired and checked by reading, not compiled, not run on hardware
//
#include <hip/hip_runtime.h>
#include <hip/hip_bf16.h>

// Mixed-precision persistent LSTM for gfx1250 (MI455X).
//  - bf16 WMMA (v_wmma_f32_16x16x32_bf16) gate GEMMs, f32 accumulate
//  - f32 cell state c lives in VGPRs across all 1024 timesteps
//  - W_ih|W_hh resident in LDS as bf16 (~65 KB of 320 KB WGP LDS)
//  - x preconverted to bf16 by a parallel kernel (if workspace allows)
//  - done-masking applied at PRODUCE time (state written at step t-1 is
//    pre-masked with done[t]) so the recurrence loop's h reads are plain
//    unconditional global_load_b128 (no cndmask'd addresses, no spills)
//  - one device-scope spin barrier per timestep across 16 co-resident WGs

typedef __attribute__((ext_vector_type(16))) __bf16 v16bf;
typedef __attribute__((ext_vector_type(8)))  float  v8f;

#define LSTM_T 1024
#define LSTM_B 64
#define LSTM_D 256
#define LSTM_H 256
#define LSTM_K (LSTM_D + LSTM_H)   // combined K = 512
#define KP     520                 // padded LDS row stride (bf16), mult of 8 -> 16B-aligned rows
#define NBLK   16                  // one workgroup per 16 hidden columns
#define NTHR   128                 // 4 waves; wave = 16 batch rows

union FragBf {
  uint4 u[2];   // two 16B chunks (8 bf16 each)
  v16bf v;      // 16 bf16 elements = 8 VGPRs
};

union Pack8 {
  uint4  u;
  __bf16 h[8];
};

__device__ __forceinline__ float sigm_f(float x) {
  return __builtin_amdgcn_rcpf(1.0f + __expf(-x));      // stable: e^-x->inf gives 0
}
__device__ __forceinline__ float tanh_f(float x) {
  return 1.0f - 2.0f * __builtin_amdgcn_rcpf(1.0f + __expf(2.0f * x));  // stable at +/-inf
}

// ---- one-shot: h0 (masked by done[0]) -> bf16 ping buffer; reset barrier ----
__global__ void lstm_init(const float* __restrict__ h0,
                          const int*   __restrict__ done,
                          __bf16* __restrict__ hbuf0,
                          unsigned* __restrict__ counter) {
  int i = blockIdx.x * blockDim.x + threadIdx.x;
  if (i < LSTM_B * LSTM_H) {
    int b = i / LSTM_H;
    hbuf0[i] = done[b] ? (__bf16)0.0f : (__bf16)h0[i];   // done at t=0
  }
  if (i == 0) *counter = 0u;
}

// ---- one-shot: x (f32, 64 MB) -> bf16 (32 MB), fully parallel, HBM-bound ----
__global__ void x_to_bf16(const float* __restrict__ xin, __bf16* __restrict__ xout) {
  size_t i = ((size_t)blockIdx.x * blockDim.x + threadIdx.x) * 8;
  float4 f0 = *(const float4*)(xin + i);
  float4 f1 = *(const float4*)(xin + i + 4);
  Pack8 p;
  p.h[0] = (__bf16)f0.x; p.h[1] = (__bf16)f0.y; p.h[2] = (__bf16)f0.z; p.h[3] = (__bf16)f0.w;
  p.h[4] = (__bf16)f1.x; p.h[5] = (__bf16)f1.y; p.h[6] = (__bf16)f1.z; p.h[7] = (__bf16)f1.w;
  *(uint4*)(xout + i) = p.u;
}

template <bool XBF>
__launch_bounds__(NTHR, 1)
__global__ void lstm_persistent(const float*  __restrict__ x,
                                const __bf16* __restrict__ xbf,
                                const float*  __restrict__ c0,
                                const float*  __restrict__ W_ih,
                                const float*  __restrict__ W_hh,
                                const float*  __restrict__ b_ih,
                                const float*  __restrict__ b_hh,
                                const int*    __restrict__ done,
                                float*        __restrict__ out,
                                __bf16*       __restrict__ hbuf,   // ping-pong 2*[B*H] bf16
                                unsigned*     __restrict__ counter) {
  __shared__ __align__(16) __bf16 Wlds[4 * 16 * KP];

  const int jt   = blockIdx.x;        // hidden-column tile (16 cols)
  const int tid  = threadIdx.x;
  const int mt   = tid >> 5;          // wave id = batch-row tile (16 rows)
  const int lane = tid & 31;
  const int half = lane >> 4;
  const int n16  = lane & 15;

  // ---- Stage W_ih|W_hh (this WG's 64 gate columns, K=512) into LDS, once ----
  for (int idx = tid; idx < 4 * 16 * LSTM_K; idx += NTHR) {
    int k = idx % LSTM_K;
    int n = (idx / LSTM_K) % 16;
    int g = idx / (LSTM_K * 16);
    int row = g * LSTM_H + jt * 16 + n;              // gate-column = W row
    float w = (k < LSTM_D) ? W_ih[row * LSTM_D + k]
                           : W_hh[row * LSTM_H + (k - LSTM_D)];
    Wlds[(g * 16 + n) * KP + k] = (__bf16)w;
  }
  __syncthreads();

  float bsum[4];
  #pragma unroll
  for (int g = 0; g < 4; ++g) {
    int col = g * LSTM_H + jt * 16 + n16;
    bsum[g] = b_ih[col] + b_hh[col];
  }

  const int rowA  = mt * 16 + n16;      // A-fragment batch row (M = lane&15)
  const int rowC0 = mt * 16 + 8 * half; // C/D-fragment row base (row = r + 8*half)

  // c state (C/D layout), pre-masked with done[0]
  v8f c_frag, h_frag;
  #pragma unroll
  for (int r = 0; r < 8; ++r) {
    int row = rowC0 + r;
    c_frag[r] = done[row] ? 0.0f : c0[(size_t)row * LSTM_H + jt * 16 + n16];
  }

  for (int t = 0; t < LSTM_T; ++t) {
    const __bf16* hb_rd = hbuf + (size_t)(t & 1) * (LSTM_B * LSTM_H);
    __bf16*       hb_wr = hbuf + (size_t)((t + 1) & 1) * (LSTM_B * LSTM_H);

    if (t + 1 < LSTM_T) {  // pull next step's x row toward the WGP (global_prefetch_b8)
      if (XBF) __builtin_prefetch(&xbf[((size_t)(t + 1) * LSTM_B + rowA) * LSTM_D], 0, 1);
      else     __builtin_prefetch(&x  [((size_t)(t + 1) * LSTM_B + rowA) * LSTM_D], 0, 1);
    }

    v8f acc[4];
    #pragma unroll
    for (int g = 0; g < 4; ++g)
      #pragma unroll
      for (int e = 0; e < 8; ++e) acc[g][e] = bsum[g];

    // K loop over combined [x | h] (K=512): kt 0..7 -> x, kt 8..15 -> h.
    // h was pre-masked when written at step t-1, so reads are unconditional.
    #pragma unroll
    for (int kt = 0; kt < 16; ++kt) {
      FragBf a;
      if (kt < 8) {
        if (XBF) {
          const __bf16* xp = xbf + ((size_t)t * LSTM_B + rowA) * LSTM_D + kt * 32 + half * 8;
          a.u[0] = *(const uint4*)(xp);
          a.u[1] = *(const uint4*)(xp + 16);
        } else {
          const float* xp = x + ((size_t)t * LSTM_B + rowA) * LSTM_D + kt * 32 + half * 8;
          float4 f0 = *(const float4*)(xp + 0);
          float4 f1 = *(const float4*)(xp + 4);
          float4 f2 = *(const float4*)(xp + 16);
          float4 f3 = *(const float4*)(xp + 20);
          a.v[0]  = (__bf16)f0.x; a.v[1]  = (__bf16)f0.y; a.v[2]  = (__bf16)f0.z; a.v[3]  = (__bf16)f0.w;
          a.v[4]  = (__bf16)f1.x; a.v[5]  = (__bf16)f1.y; a.v[6]  = (__bf16)f1.z; a.v[7]  = (__bf16)f1.w;
          a.v[8]  = (__bf16)f2.x; a.v[9]  = (__bf16)f2.y; a.v[10] = (__bf16)f2.z; a.v[11] = (__bf16)f2.w;
          a.v[12] = (__bf16)f3.x; a.v[13] = (__bf16)f3.y; a.v[14] = (__bf16)f3.z; a.v[15] = (__bf16)f3.w;
        }
      } else {
        const __bf16* hp = hb_rd + (size_t)rowA * LSTM_H + (kt - 8) * 32 + half * 8;
        a.u[0] = *(const uint4*)(hp);
        a.u[1] = *(const uint4*)(hp + 16);
      }
      #pragma unroll
      for (int g = 0; g < 4; ++g) {
        FragBf b;  // B 32x16 bf16: lane = col n16; K = 8*half+{0..7}, 16+8*half+{0..7}
        const __bf16* wp = &Wlds[(g * 16 + n16) * KP + kt * 32 + half * 8];
        b.u[0] = *(const uint4*)(wp);
        b.u[1] = *(const uint4*)(wp + 16);
        acc[g] = __builtin_amdgcn_wmma_f32_16x16x32_bf16(
            false, a.v, false, b.v, (short)0, acc[g], false, false);
      }
    }

    // Next-step done mask (binary) — used to pre-mask the state we WRITE.
    // At t+1 == T no masking (returned hT/cT are the raw last states).
    int dmn[8];
    if (t + 1 < LSTM_T) {
      const int4* dp = (const int4*)&done[(t + 1) * LSTM_B + rowC0];
      int4 d0 = dp[0], d1 = dp[1];
      dmn[0] = d0.x; dmn[1] = d0.y; dmn[2] = d0.z; dmn[3] = d0.w;
      dmn[4] = d1.x; dmn[5] = d1.y; dmn[6] = d1.z; dmn[7] = d1.w;
    } else {
      #pragma unroll
      for (int r = 0; r < 8; ++r) dmn[r] = 0;
    }

    // LSTM cell (f32, elementwise on C/D fragment) — torch gate order i,f,g,o.
    // c_frag entering here is already masked for step t.
    #pragma unroll
    for (int r = 0; r < 8; ++r) {
      float ig = sigm_f(acc[0][r]);
      float fg = sigm_f(acc[1][r]);
      float gg = tanh_f(acc[2][r]);
      float og = sigm_f(acc[3][r]);
      float cn = fg * c_frag[r] + ig * gg;
      h_frag[r] = og * tanh_f(cn);      // unmasked: goes to out / hT
      c_frag[r] = dmn[r] ? 0.0f : cn;   // pre-masked for step t+1
    }

    // h out: f32 stream output (non-temporal, never re-read) + pre-masked bf16
    float* outp = out + (size_t)t * LSTM_B * LSTM_H;
    #pragma unroll
    for (int r = 0; r < 8; ++r) {
      size_t off = (size_t)(rowC0 + r) * LSTM_H + jt * 16 + n16;
      __builtin_nontemporal_store(h_frag[r], &outp[off]);
      float hm = dmn[r] ? 0.0f : h_frag[r];
      hb_wr[off] = (__bf16)hm;
    }

    // Grid-wide barrier: 16 workgroups, trivially co-resident on MI455X
    __syncthreads();
    if (tid == 0) {
      __threadfence();                                   // release h stores
      atomicAdd(counter, 1u);
      unsigned target = (unsigned)(NBLK * (t + 1));
      while (*(volatile unsigned*)counter < target) {
        __builtin_amdgcn_s_sleep(1);
      }
    }
    __syncthreads();
    __threadfence();                                     // acquire other WGs' h
  }

  // Final states hT, cT appended after the [T,B,H] output (unmasked)
  float* hT = out + (size_t)LSTM_T * LSTM_B * LSTM_H;
  float* cT = hT + LSTM_B * LSTM_H;
  #pragma unroll
  for (int r = 0; r < 8; ++r) {
    size_t off = (size_t)(rowC0 + r) * LSTM_H + jt * 16 + n16;
    hT[off] = h_frag[r];
    cT[off] = c_frag[r];
  }
}

extern "C" void kernel_launch(void* const* d_in, const int* in_sizes, int n_in,
                              void* d_out, int out_size, void* d_ws, size_t ws_size,
                              hipStream_t stream) {
  const float* x    = (const float*)d_in[0];
  const float* h0   = (const float*)d_in[1];
  const float* c0   = (const float*)d_in[2];
  const float* W_ih = (const float*)d_in[3];
  const float* W_hh = (const float*)d_in[4];
  const float* b_ih = (const float*)d_in[5];
  const float* b_hh = (const float*)d_in[6];
  const int*   done = (const int*)d_in[7];
  // d_in[8] = bptt_len: forward no-op

  float* out = (float*)d_out;

  // Workspace layout:
  //   [0..4)        barrier counter
  //   [256..64K+)   ping-pong bf16 h buffers (2 * B*H * 2B = 64 KB)
  //   [1MB..+32MB)  optional bf16 copy of x (fragment-consumable)
  unsigned* counter = (unsigned*)d_ws;
  __bf16*   hbuf    = (__bf16*)((char*)d_ws + 256);

  const size_t xbf_off   = (size_t)1 << 20;
  const size_t xbf_bytes = (size_t)LSTM_T * LSTM_B * LSTM_D * 2;
  __bf16* xbf = (__bf16*)((char*)d_ws + xbf_off);
  const bool use_xbf = ws_size >= xbf_off + xbf_bytes;   // ws_size fixed per run -> deterministic

  lstm_init<<<(LSTM_B * LSTM_H + 255) / 256, 256, 0, stream>>>(h0, done, hbuf, counter);

  if (use_xbf) {
    const int n8 = LSTM_T * LSTM_B * LSTM_D / 8;         // 8 elems per thread
    x_to_bf16<<<n8 / 256, 256, 0, stream>>>(x, xbf);
    lstm_persistent<true><<<NBLK, NTHR, 0, stream>>>(x, xbf, c0, W_ih, W_hh,
                                                     b_ih, b_hh, done, out, hbuf, counter);
  } else {
    lstm_persistent<false><<<NBLK, NTHR, 0, stream>>>(x, xbf, c0, W_ih, W_hh,
                                                      b_ih, b_hh, done, out, hbuf, counter);
  }
}
